// GVPEncoder_68624987455944
// MI455X (gfx1250) — matrix operand order repaired and checked by reading, not compiled, not run on hardware
//
#include <hip/hip_runtime.h>
#include <math.h>

typedef _Float16 v16h __attribute__((ext_vector_type(16)));
typedef _Float16 h8   __attribute__((ext_vector_type(8)));
typedef float    v8f  __attribute__((ext_vector_type(8)));

#define HD 128
#define NLAYERS 3

// ---- monotone float <-> uint mapping for atomic segment-max ----
__device__ __forceinline__ unsigned int fmap_ord(float f) {
  unsigned int u = __float_as_uint(f);
  return (u & 0x80000000u) ? ~u : (u | 0x80000000u);
}
__device__ __forceinline__ float funmap_ord(unsigned int u) {
  unsigned int v = (u & 0x80000000u) ? (u & 0x7fffffffu) : ~u;
  return __uint_as_float(v);
}

__global__ void k_zero_f32(float* __restrict__ p, int n) {
  int i = blockIdx.x * blockDim.x + threadIdx.x;
  if (i < n) p[i] = 0.f;
}

// f32 -> f16 elementwise shadow copy
__global__ void k_cvt_f16(const float* __restrict__ in, _Float16* __restrict__ out, int n) {
  int i = blockIdx.x * blockDim.x + threadIdx.x;
  if (i < n) out[i] = (_Float16)in[i];
}

// W [K,128] f32 -> Wt [128,K] f16 (transpose + convert, done once per launch)
__global__ void k_wt16(const float* __restrict__ W, _Float16* __restrict__ Wt, int K) {
  int i = blockIdx.x * blockDim.x + threadIdx.x;
  if (i >= K * HD) return;
  int k = i / HD, n = i % HD;
  Wt[(size_t)n * K + k] = (_Float16)W[i];
}

// per-edge: distance feature + degree / dist-sum accumulation (mean self-loop attr)
__global__ void k_edge_pre(const int* __restrict__ src, const int* __restrict__ dst,
                           const float* __restrict__ pos,
                           float* __restrict__ dist, float* __restrict__ deg,
                           float* __restrict__ sumd, int E) {
  int e = blockIdx.x * blockDim.x + threadIdx.x;
  if (e >= E) return;
  int s = src[e], d = dst[e];
  float dx = pos[3 * s    ] - pos[3 * d    ];
  float dy = pos[3 * s + 1] - pos[3 * d + 1];
  float dz = pos[3 * s + 2] - pos[3 * d + 2];
  float r = sqrtf(dx * dx + dy * dy + dz * dz + 1e-12f);
  dist[e] = r;
  atomicAdd(&deg[d], 1.0f);
  atomicAdd(&sumd[d], r);
}

__global__ void k_loop_attr(const float* __restrict__ deg, const float* __restrict__ sumd,
                            float* __restrict__ loop_attr, int n) {
  int i = blockIdx.x * blockDim.x + threadIdx.x;
  if (i >= n) return;
  loop_attr[i] = sumd[i] / fmaxf(deg[i], 1.0f);
}

// ---------------------------------------------------------------------------
// WMMA GEMM, all-f16 operands: C = A16[M,K] @ B + bias (B given transposed,
// f16 [128,K]).  One wave computes a 16x64 strip: one A fragment reused across
// 4 B fragments / 4 accumulators; K templated so the k-loop fully unrolls
// (K=128 -> 16 v_wmma inlined).  Fragments per CDNA5 ISA 7.12.2 (wave32):
//   A 16x32: lanes 0-15 hold K {0..7,16..23}, lanes 16-31 {8..15,24..31}
//   B 32x16: lanes 0-15 (N=lane) hold K {0..7,16..23}, lanes 16-31 {8..15,24..31}
//   C/D:     lanes 0-15 N=lane M=r, lanes 16-31 N=lane-16 M=8+r
// Strips [0,nsA) -> (Bt0,bias0,C0[,C16]) ; [nsA,nsA+nsB) -> (Bt1,bias1,C1).
// ---------------------------------------------------------------------------
template <int K>
__global__ void k_wmma_gemm(const _Float16* __restrict__ A16,
                            const _Float16* __restrict__ Bt0,
                            const float* __restrict__ bias0,
                            float* __restrict__ C0, _Float16* __restrict__ C16,
                            const _Float16* __restrict__ Bt1,
                            const float* __restrict__ bias1, float* __restrict__ C1,
                            int M, int nsA, int nsB, int clampAct) {
  int wave = (blockIdx.x * blockDim.x + threadIdx.x) >> 5;
  int lane = threadIdx.x & 31;
  int nstot = nsA + nsB;
  int mt = wave / nstot;
  int ns = wave % nstot;
  if (mt >= (M >> 4)) return;  // wave-uniform: EXEC stays all-ones for WMMA

  const _Float16* Bt; const float* bias; float* C; _Float16* Csh; int nbase;
  if (ns < nsA) { Bt = Bt0; bias = bias0; C = C0; Csh = C16; nbase = ns * 64; }
  else          { Bt = Bt1; bias = bias1; C = C1; Csh = nullptr; nbase = (ns - nsA) * 64; }

  int lrow  = lane & 15;
  int khalf = lane >> 4;
  int m0 = mt << 4;

  const h8* arow = (const h8*)(A16 + (size_t)(m0 + lrow) * K);  // K/8 h8 per row
  v8f acc[4];
#pragma unroll
  for (int t = 0; t < 4; ++t) acc[t] = (v8f){0.f, 0.f, 0.f, 0.f, 0.f, 0.f, 0.f, 0.f};

#pragma unroll
  for (int k0 = 0; k0 < K; k0 += 32) {
    // A fragment: two 16B loads
    int ia = (k0 >> 3) + khalf;
    h8 alo = arow[ia];
    h8 ahi = arow[ia + 2];
    v16h a = __builtin_shufflevector(alo, ahi, 0, 1, 2, 3, 4, 5, 6, 7,
                                     8, 9, 10, 11, 12, 13, 14, 15);
#pragma unroll
    for (int t = 0; t < 4; ++t) {
      int n = nbase + t * 16 + lrow;
      const h8* brow = (const h8*)(Bt + (size_t)n * K);
      int ib = (k0 >> 3) + khalf;
      h8 blo = brow[ib];
      h8 bhi = brow[ib + 2];
      v16h b = __builtin_shufflevector(blo, bhi, 0, 1, 2, 3, 4, 5, 6, 7,
                                       8, 9, 10, 11, 12, 13, 14, 15);
      acc[t] = __builtin_amdgcn_wmma_f32_16x16x32_f16(false, a, false, b,
                                                      (short)0, acc[t], false, false);
    }
  }

  int rbase = m0 + khalf * 8;
#pragma unroll
  for (int t = 0; t < 4; ++t) {
    int n = nbase + t * 16 + lrow;
    float bv = bias[n];
#pragma unroll
    for (int r = 0; r < 8; ++r) {
      float v = acc[t][r] + bv;
      if (clampAct) v = fminf(fmaxf(v, -10.f), 10.f);
      size_t idx = (size_t)(rbase + r) * HD + n;
      C[idx] = v;
      if (Csh) Csh[idx] = (_Float16)v;
    }
  }
}

// zero per-layer accumulators (segmax/denom/outacc)
__global__ void k_layer_init(unsigned int* __restrict__ segmax, float* __restrict__ denom,
                             float* __restrict__ outacc, int n) {
  int i = blockIdx.x * blockDim.x + threadIdx.x;
  if (i < n * HD) outacc[i] = 0.f;
  if (i < n) { segmax[i] = 0u; denom[i] = 0.f; }
}

// wave-per-edge, float4 per lane: msg = xl[src]+xr[dst]+attr*We ; leaky_relu ;
// logit = g . att ; atomic segment-max on dst
__global__ void k_edge_logits(const float* __restrict__ xl, const float* __restrict__ xr,
                              const int* __restrict__ src, const int* __restrict__ dst,
                              const float* __restrict__ dist, const float* __restrict__ loop_attr,
                              const float* __restrict__ We, const float* __restrict__ att,
                              float* __restrict__ logits, unsigned int* __restrict__ segmax,
                              int E, int n) {
  int wid  = (blockIdx.x * blockDim.x + threadIdx.x) >> 5;
  int lane = threadIdx.x & 31;
  int total = E + n;
  if (wid >= total) return;
  int s, d; float attr;
  if (wid < E) { s = src[wid]; d = dst[wid]; attr = dist[wid]; }
  else         { s = d = wid - E; attr = loop_attr[wid - E]; }
  const float4 l  = ((const float4*)(xl + (size_t)s * HD))[lane];
  const float4 r  = ((const float4*)(xr + (size_t)d * HD))[lane];
  const float4 w  = ((const float4*)We)[lane];
  const float4 aw = ((const float4*)att)[lane];
  float m0 = l.x + r.x + attr * w.x;
  float m1 = l.y + r.y + attr * w.y;
  float m2 = l.z + r.z + attr * w.z;
  float m3 = l.w + r.w + attr * w.w;
  float acc = (m0 > 0.f ? m0 : 0.2f * m0) * aw.x
            + (m1 > 0.f ? m1 : 0.2f * m1) * aw.y
            + (m2 > 0.f ? m2 : 0.2f * m2) * aw.z
            + (m3 > 0.f ? m3 : 0.2f * m3) * aw.w;
#pragma unroll
  for (int off = 16; off > 0; off >>= 1) acc += __shfl_xor(acc, off, 32);
  if (lane == 0) {
    logits[wid] = acc;
    atomicMax(&segmax[d], fmap_ord(acc));
  }
}

// thread-per-edge: ex = exp(logit - segmax[dst]); accumulate denom
__global__ void k_edge_exp(const int* __restrict__ dst,
                           const unsigned int* __restrict__ segmax,
                           float* __restrict__ logits, float* __restrict__ denom,
                           int E, int n) {
  int e = blockIdx.x * blockDim.x + threadIdx.x;
  int total = E + n;
  if (e >= total) return;
  int d = (e < E) ? dst[e] : (e - E);
  float m = funmap_ord(segmax[d]);
  float ex = __expf(logits[e] - m);
  logits[e] = ex;  // overwrite in place
  atomicAdd(&denom[d], ex);
}

// wave-per-edge, float4 per lane: out[dst] += alpha * xl[src]
__global__ void k_edge_scatter(const float* __restrict__ xl,
                               const int* __restrict__ src, const int* __restrict__ dst,
                               const float* __restrict__ logits, const float* __restrict__ denom,
                               float* __restrict__ outacc, int E, int n) {
  int wid  = (blockIdx.x * blockDim.x + threadIdx.x) >> 5;
  int lane = threadIdx.x & 31;
  int total = E + n;
  if (wid >= total) return;
  int s, d;
  if (wid < E) { s = src[wid]; d = dst[wid]; }
  else         { s = d = wid - E; }
  float alpha = logits[wid] / fmaxf(denom[d], 1e-16f);
  const float4 l = ((const float4*)(xl + (size_t)s * HD))[lane];
  float* po = outacc + (size_t)d * HD + 4 * lane;
  atomicAdd(po + 0, alpha * l.x);
  atomicAdd(po + 1, alpha * l.y);
  atomicAdd(po + 2, alpha * l.z);
  atomicAdd(po + 3, alpha * l.w);
}

// s = relu(clip(s + out + bias, -20, 20)); also refresh f16 shadow of s
__global__ void k_epilogue(float* __restrict__ s, _Float16* __restrict__ s16,
                           const float* __restrict__ outacc,
                           const float* __restrict__ bias, int n) {
  int i = blockIdx.x * blockDim.x + threadIdx.x;
  if (i >= n * HD) return;
  float v = s[i] + outacc[i] + bias[i & (HD - 1)];
  v = fminf(fmaxf(v, -20.f), 20.f);
  v = fmaxf(v, 0.f);
  s[i] = v;
  s16[i] = (_Float16)v;
}

extern "C" void kernel_launch(void* const* d_in, const int* in_sizes, int n_in,
                              void* d_out, int out_size, void* d_ws, size_t ws_size,
                              hipStream_t stream) {
  const float* x     = (const float*)d_in[0];
  const int*   ei    = (const int*)  d_in[1];
  const float* pos   = (const float*)d_in[2];
  const float* W_emb = (const float*)d_in[3];
  const float* b_emb = (const float*)d_in[4];
  const float* Wl    = (const float*)d_in[5];
  const float* bl    = (const float*)d_in[6];
  const float* Wr    = (const float*)d_in[7];
  const float* br    = (const float*)d_in[8];
  const float* We    = (const float*)d_in[9];
  const float* att   = (const float*)d_in[10];
  const float* bias  = (const float*)d_in[11];

  int N = in_sizes[0] / 64;  // IN = 64
  int E = in_sizes[1] / 2;
  const int* src = ei;
  const int* dst = ei + E;

  float* s = (float*)d_out;           // [N,128], computed in place
  float* v = s + (size_t)N * HD;      // [N,16,3] -> zeros

  // ---- workspace layout ----
  float* ws = (float*)d_ws;
  size_t off = 0;
  auto alignup = [&]() { off = (off + 3) & ~(size_t)3; };  // 16B alignment
  float* xl        = ws + off; off += (size_t)N * HD;
  float* xr        = ws + off; off += (size_t)N * HD;
  float* outacc    = ws + off; off += (size_t)N * HD;
  float* dist      = ws + off; off += (size_t)E;      alignup();
  float* logits    = ws + off; off += (size_t)E + N;  alignup();
  float* loop_attr = ws + off; off += N;              alignup();
  float* deg       = ws + off; off += N;              alignup();
  float* sumd      = ws + off; off += N;              alignup();
  float* denom     = ws + off; off += N;              alignup();
  unsigned int* segmax = (unsigned int*)(ws + off); off += N; alignup();
  // f16 region (sizes in halves -> floats/2, keep 16B alignment)
  _Float16* x16    = (_Float16*)(ws + off); off += (size_t)N * 64 / 2;  alignup();
  _Float16* s16    = (_Float16*)(ws + off); off += (size_t)N * HD / 2; alignup();
  _Float16* wembT  = (_Float16*)(ws + off); off += (size_t)64 * HD / 2; alignup();
  _Float16* wlT    = (_Float16*)(ws + off); off += (size_t)NLAYERS * HD * HD / 2; alignup();
  _Float16* wrT    = (_Float16*)(ws + off); off += (size_t)NLAYERS * HD * HD / 2; alignup();

  int total = E + N;
  dim3 blk(256);
  auto cdiv = [](long a, long b) { return (int)((a + b - 1) / b); };

  // --- one-time conversions (every call; deterministic) ---
  k_cvt_f16<<<cdiv((long)N * 64, 256), blk, 0, stream>>>(x, x16, N * 64);
  k_wt16<<<cdiv(64 * HD, 256), blk, 0, stream>>>(W_emb, wembT, 64);
  for (int l = 0; l < NLAYERS; ++l) {
    k_wt16<<<cdiv(HD * HD, 256), blk, 0, stream>>>(Wl + (size_t)l * HD * HD,
                                                   wlT + (size_t)l * HD * HD, HD);
    k_wt16<<<cdiv(HD * HD, 256), blk, 0, stream>>>(Wr + (size_t)l * HD * HD,
                                                   wrT + (size_t)l * HD * HD, HD);
  }

  // --- precompute edge features ---
  k_zero_f32<<<cdiv(N, 256), blk, 0, stream>>>(deg, N);
  k_zero_f32<<<cdiv(N, 256), blk, 0, stream>>>(sumd, N);
  k_zero_f32<<<cdiv((long)N * 48, 256), blk, 0, stream>>>(v, N * 48);
  k_edge_pre<<<cdiv(E, 256), blk, 0, stream>>>(src, dst, pos, dist, deg, sumd, E);
  k_loop_attr<<<cdiv(N, 256), blk, 0, stream>>>(deg, sumd, loop_attr, N);

  // --- embedding GEMM: s = clip(x @ W_emb + b_emb, +-10), also write s16 ---
  {
    long waves = (long)(N / 16) * 2;  // 2 strips of 64 columns
    k_wmma_gemm<64><<<cdiv(waves * 32, 256), blk, 0, stream>>>(
        x16, wembT, b_emb, s, s16, nullptr, nullptr, nullptr, N, 2, 0, 1);
  }

  // --- layers ---
  for (int l = 0; l < NLAYERS; ++l) {
    long waves = (long)(N / 16) * 4;  // xl: 2 strips + xr: 2 strips
    k_wmma_gemm<128><<<cdiv(waves * 32, 256), blk, 0, stream>>>(
        s16, wlT + (size_t)l * HD * HD, bl + (size_t)l * HD, xl, nullptr,
             wrT + (size_t)l * HD * HD, br + (size_t)l * HD, xr, N, 2, 2, 0);

    k_layer_init<<<cdiv((long)N * HD, 256), blk, 0, stream>>>(segmax, denom, outacc, N);

    k_edge_logits<<<cdiv((long)total * 32, 256), blk, 0, stream>>>(
        xl, xr, src, dst, dist, loop_attr, We + (size_t)l * HD, att + (size_t)l * HD,
        logits, segmax, E, N);

    k_edge_exp<<<cdiv(total, 256), blk, 0, stream>>>(dst, segmax, logits, denom, E, N);

    k_edge_scatter<<<cdiv((long)total * 32, 256), blk, 0, stream>>>(
        xl, src, dst, logits, denom, outacc, E, N);

    k_epilogue<<<cdiv((long)N * HD, 256), blk, 0, stream>>>(
        s, s16, outacc, bias + (size_t)l * HD, N);
  }
}
